// SelfAttention_84327387890487
// MI455X (gfx1250) — matrix operand order repaired and checked by reading, not compiled, not run on hardware
//
#include <hip/hip_runtime.h>
#include <math.h>
#include <stdint.h>

#define BATCH 4
#define CH    128
#define NTOK  4096
#define GRP   8
#define CPG   16        // CH / GRP
#define EPSV  1e-5f
#define QSCALE 0.08838834764831845f   // 1/sqrt(128)

typedef __attribute__((ext_vector_type(16))) _Float16 v16h;
typedef __attribute__((ext_vector_type(8)))  _Float16 v8h;
typedef __attribute__((ext_vector_type(8)))  float    v8f;
typedef __attribute__((ext_vector_type(4)))  unsigned int v4u;
typedef __attribute__((ext_vector_type(8)))  int      v8i_;
typedef __attribute__((ext_vector_type(4)))  int      v4i_;

union V16H { v16h v; v8h h[2]; _Float16 e[16]; };
union V8H  { v8h  v; _Float16 e[8]; };
union V8F  { v8f  v; float e[8]; };

// ---- WMMA fragment helpers (contiguous K = 16*(lane/16) + e in A and B, wave32) ----------

// A-matrix 16x32 f16, src row-major [16][ld]: lane holds row (lane&15),
// 16 contiguous K at k0=(lane<16?0:16).
__device__ __forceinline__ v16h load_a_frag(const _Float16* src, int ld) {
  const int lane = threadIdx.x & 31;
  const int m  = lane & 15;
  const int k0 = (lane & 16) ? 16 : 0;
  const _Float16* p = src + (size_t)m * ld + k0;
  V16H r;
  r.h[0] = *(const v8h*)(p);        // K = k0 .. k0+7
  r.h[1] = *(const v8h*)(p + 8);    // K = k0+8 .. k0+15
  return r.v;
}

// B-matrix 32x16 f16 where B[k][n] is read from row-major R[n][k] (ld = row
// stride of R): lane holds column n=(lane&15), K = k0 + e, k0=(lane<16?0:16).
__device__ __forceinline__ v16h load_bt_frag(const _Float16* R, int ld) {
  const int lane = threadIdx.x & 31;
  const int n  = lane & 15;
  const int k0 = (lane & 16) ? 16 : 0;
  return *(const v16h*)(R + (size_t)n * ld + k0);
}

__device__ __forceinline__ v8f wmma16(v16h a, v16h b, v8f c) {
  return __builtin_amdgcn_wmma_f32_16x16x32_f16(false, a, false, b, (short)0, c, false, false);
}

// ---- Tensor Data Mover: 2D tile load Global->LDS (D# per ISA 08, §8.3/8.4) -
// Tensor: rows of tensor_d0 elements (2-byte), row stride stride0 elements.
// Tile: tile_d1 rows x tile_d0 elements, packed row-major into LDS at lds_off.
__device__ __forceinline__ void tdm_load_2d(const _Float16* gsrc, unsigned lds_off,
                                            unsigned tensor_d0, unsigned tensor_d1,
                                            unsigned tile_d0, unsigned tile_d1,
                                            unsigned long long stride0) {
  unsigned long long ga = (unsigned long long)(uintptr_t)gsrc;
  v4u g0;
  g0[0] = 1u;                                          // count=1, user, no gather
  g0[1] = lds_off;                                     // lds_addr (bytes)
  g0[2] = (unsigned)(ga & 0xFFFFFFFFu);                // global_addr[31:0]
  g0[3] = (unsigned)((ga >> 32) & 0x01FFFFFFu)         // global_addr[56:32]
        | (2u << 30);                                  // type = 2 ("image")
  v8i_ g1;
  g1[0] = (int)(1u << 16);                             // data_size=1 (2 bytes)
  g1[1] = (int)((tensor_d0 & 0xFFFFu) << 16);          // tensor_dim0[15:0]
  g1[2] = (int)(((tensor_d0 >> 16) & 0xFFFFu) |        // tensor_dim0[31:16]
                ((tensor_d1 & 0xFFFFu) << 16));        // tensor_dim1[15:0]
  g1[3] = (int)(((tensor_d1 >> 16) & 0xFFFFu) |        // tensor_dim1[31:16]
                ((tile_d0 & 0xFFFFu) << 16));          // tile_dim0
  g1[4] = (int)(tile_d1 & 0xFFFFu);                    // tile_dim1 (tile_dim2=0)
  g1[5] = (int)(unsigned)(stride0 & 0xFFFFFFFFu);      // tensor_dim0_stride[31:0]
  g1[6] = (int)((stride0 >> 32) & 0xFFFFu);            // stride[47:32], dim1_stride=0
  g1[7] = 0;
  v4i_ z4 = {0, 0, 0, 0};
#if defined(__clang_major__) && (__clang_major__ >= 23)
  v8i_ z8 = {0, 0, 0, 0, 0, 0, 0, 0};
  __builtin_amdgcn_tensor_load_to_lds(g0, g1, z4, z4, z8, 0);
#else
  __builtin_amdgcn_tensor_load_to_lds(g0, g1, z4, z4, 0);
#endif
}

// ---- Kernel 1: GroupNorm statistics (one block per (b,g)) ------------------

__global__ void gn_stats_kernel(const float* __restrict__ x, float* __restrict__ stats) {
  const int bg = blockIdx.x;                 // 0..31
  const int b = bg / GRP, g = bg % GRP;
  const float* base = x + ((size_t)b * CH + (size_t)g * CPG) * NTOK;
  float s = 0.f, ss = 0.f;
  for (int i = threadIdx.x; i < CPG * NTOK; i += blockDim.x) {
    float v = base[i];
    s += v; ss += v * v;
  }
  __shared__ float sh_s[256], sh_ss[256];
  sh_s[threadIdx.x] = s; sh_ss[threadIdx.x] = ss;
  __syncthreads();
  for (int off = 128; off > 0; off >>= 1) {
    if ((int)threadIdx.x < off) {
      sh_s[threadIdx.x]  += sh_s[threadIdx.x + off];
      sh_ss[threadIdx.x] += sh_ss[threadIdx.x + off];
    }
    __syncthreads();
  }
  if (threadIdx.x == 0) {
    const float inv_n = 1.0f / (float)(CPG * NTOK);
    float mean = sh_s[0] * inv_n;
    float var  = sh_ss[0] * inv_n - mean * mean;
    stats[bg * 2 + 0] = mean;
    stats[bg * 2 + 1] = rsqrtf(var + EPSV);
  }
}

// ---- Kernel 2: normalize + affine + f16 convert, [B][C][N] -> [B][N][C] ----

__global__ void gn_apply_kernel(const float* __restrict__ x,
                                const float* __restrict__ gw,
                                const float* __restrict__ gb,
                                const float* __restrict__ stats,
                                _Float16* __restrict__ h16) {
  size_t idx = (size_t)blockIdx.x * blockDim.x + threadIdx.x;   // over B*CH*NTOK
  const int n = (int)(idx % NTOK);
  const int c = (int)((idx / NTOK) % CH);
  const int b = (int)(idx / ((size_t)NTOK * CH));
  const int bg = b * GRP + c / CPG;
  const float mean = stats[bg * 2 + 0];
  const float rstd = stats[bg * 2 + 1];
  const float v = (x[idx] - mean) * rstd * gw[c] + gb[c];
  h16[((size_t)b * NTOK + n) * CH + c] = (_Float16)v;
}

// ---- Kernel 3: weights f32 -> f16 ------------------------------------------

__global__ void cvt_w_kernel(const float* __restrict__ qkv_w,
                             const float* __restrict__ proj_w,
                             _Float16* __restrict__ w16q,
                             _Float16* __restrict__ w16p) {
  const int i = blockIdx.x * blockDim.x + threadIdx.x;          // 0 .. 3*CH*CH-1
  w16q[i] = (_Float16)qkv_w[i];
  if (i < CH * CH) w16p[i] = (_Float16)proj_w[i];
}

// ---- Kernel 4: QKV GEMM (one 16x16 tile per wave) --------------------------
// q16,k16: [B][N][C] f16 (q pre-scaled by 1/sqrt(C)); v16: [B][C][N] f16.

__global__ void __launch_bounds__(256) qkv_gemm_kernel(const _Float16* __restrict__ h16,
                                                       const _Float16* __restrict__ w16q,
                                                       const float* __restrict__ qkv_b,
                                                       _Float16* __restrict__ q16,
                                                       _Float16* __restrict__ k16,
                                                       _Float16* __restrict__ v16) {
  const int gw = blockIdx.x * (blockDim.x >> 5) + (threadIdx.x >> 5);
  const int ot = gw % 24;                 // 384/16 output-col tiles
  const int mt = gw / 24;                 // B*N/16 row tiles
  const int b  = mt >> 8;
  const int n0 = (mt & 255) << 4;
  const int o0 = ot << 4;
  const int lane = threadIdx.x & 31;
  const int nloc = lane & 15;
  const int hi   = (lane >> 4) & 1;

  const _Float16* abase = h16 + ((size_t)b * NTOK + n0) * CH;
  v8f acc = {};
  for (int kk = 0; kk < 4; ++kk) {
    v16h a  = load_a_frag(abase + kk * 32, CH);
    v16h bf = load_bt_frag(w16q + (size_t)o0 * CH + kk * 32, CH);
    acc = wmma16(a, bf, acc);
  }
  const int o = o0 + nloc;
  const float bias = qkv_b[o];
  V8F d; d.v = acc;

  if (o < CH) {                                    // Q, scaled
    _Float16* dst = q16 + (size_t)b * NTOK * CH + o;
    for (int r = 0; r < 8; ++r) {
      int n = n0 + r + 8 * hi;
      dst[(size_t)n * CH] = (_Float16)((d.e[r] + bias) * QSCALE);
    }
  } else if (o < 2 * CH) {                         // K
    const int c = o - CH;
    _Float16* dst = k16 + (size_t)b * NTOK * CH + c;
    for (int r = 0; r < 8; ++r) {
      int n = n0 + r + 8 * hi;
      dst[(size_t)n * CH] = (_Float16)(d.e[r] + bias);
    }
  } else {                                         // V, transposed [C][N]
    const int c = o - 2 * CH;
    V8H pk;
    for (int r = 0; r < 8; ++r) pk.e[r] = (_Float16)(d.e[r] + bias);
    *(v8h*)(v16 + ((size_t)b * CH + c) * NTOK + n0 + 8 * hi) = pk.v;
  }
}

// ---- Kernel 5: flash attention ---------------------------------------------
// 8 waves/block share one batch; each wave owns a 16-row query tile. K/V
// 32-key chunks are TDM-DMA'd into double-buffered LDS by wave 0 and shared.

__global__ void __launch_bounds__(256) attn_kernel(const _Float16* __restrict__ q16,
                                                   const _Float16* __restrict__ k16,
                                                   const _Float16* __restrict__ v16,
                                                   _Float16* __restrict__ o16) {
  __shared__ __align__(128) _Float16 lds_k[2][32 * CH];   // [j][c], 8KB each
  __shared__ __align__(128) _Float16 lds_v[2][CH * 32];   // [c][j], 8KB each
  __shared__ __align__(64)  _Float16 lds_p[8][16 * 32];   // per-wave P stage

  const int wid  = threadIdx.x >> 5;
  const int b    = blockIdx.x >> 5;                        // 32 blocks per batch
  const int n0   = ((blockIdx.x & 31) << 7) + (wid << 4);  // 128 queries/block
  const int lane = threadIdx.x & 31;
  const int nloc = lane & 15;
  const int hi   = lane >> 4;

  const _Float16* qbase = q16 + ((size_t)b * NTOK + n0) * CH;
  const _Float16* kbase = k16 + (size_t)b * NTOK * CH;
  const _Float16* vbase = v16 + (size_t)b * CH * NTOK;
  _Float16* pl = &lds_p[wid][0];

  v16h qf[4];
  for (int kk = 0; kk < 4; ++kk) qf[kk] = load_a_frag(qbase + kk * 32, CH);

  const v8f vzero = {};
  v8f oacc[8];
  for (int t = 0; t < 8; ++t) oacc[t] = vzero;
  float mrow[8], lrow[8];
  for (int r = 0; r < 8; ++r) { mrow[r] = -1e30f; lrow[r] = 0.f; }

  // prologue: DMA chunk 0 into buffer 0
  if (wid == 0) {
    tdm_load_2d(kbase, (unsigned)(uintptr_t)&lds_k[0][0], CH, NTOK, CH, 32, CH);
    tdm_load_2d(vbase, (unsigned)(uintptr_t)&lds_v[0][0], NTOK, CH, 32, CH, NTOK);
    __builtin_amdgcn_s_wait_tensorcnt(0);
  }
  __syncthreads();

  for (int j0 = 0; j0 < NTOK; j0 += 32) {
    const int cur = (j0 >> 5) & 1;

    // DMA next chunk into the other buffer (its readers finished last iter)
    if (wid == 0 && (j0 + 32) < NTOK) {
      tdm_load_2d(kbase + (size_t)(j0 + 32) * CH,
                  (unsigned)(uintptr_t)&lds_k[cur ^ 1][0], CH, NTOK, CH, 32, CH);
      tdm_load_2d(vbase + (j0 + 32),
                  (unsigned)(uintptr_t)&lds_v[cur ^ 1][0], NTOK, CH, 32, CH, NTOK);
    }

    const _Float16* kt = &lds_k[cur][0];
    const _Float16* vt = &lds_v[cur][0];

    // S = Q @ K^T for 32 keys: two 16x16 tiles, K-dim = 128
    v8f s0 = {}, s1 = {};
    for (int kk = 0; kk < 4; ++kk) {
      v16h b0 = load_bt_frag(kt + kk * 32, CH);
      v16h b1 = load_bt_frag(kt + 16 * CH + kk * 32, CH);
      s0 = wmma16(qf[kk], b0, s0);
      s1 = wmma16(qf[kk], b1, s1);
    }
    V8F u0, u1; u0.v = s0; u1.v = s1;

    // online softmax; row m = r + 8*hi lives across a 16-lane half-wave
    float alpha[8];
    for (int r = 0; r < 8; ++r) {
      float mx = fmaxf(u0.e[r], u1.e[r]);
      mx = fmaxf(mx, __shfl_xor(mx, 1));
      mx = fmaxf(mx, __shfl_xor(mx, 2));
      mx = fmaxf(mx, __shfl_xor(mx, 4));
      mx = fmaxf(mx, __shfl_xor(mx, 8));
      const float mn = fmaxf(mrow[r], mx);
      alpha[r] = __expf(mrow[r] - mn);
      mrow[r] = mn;
      const float p0 = __expf(u0.e[r] - mn);
      const float p1 = __expf(u1.e[r] - mn);
      u0.e[r] = p0; u1.e[r] = p1;
      float ps = p0 + p1;
      ps += __shfl_xor(ps, 1);
      ps += __shfl_xor(ps, 2);
      ps += __shfl_xor(ps, 4);
      ps += __shfl_xor(ps, 8);
      lrow[r] = lrow[r] * alpha[r] + ps;
    }

    // stage P (D-layout) through LDS into A-fragment layout
    for (int r = 0; r < 8; ++r) {
      const int row = r + 8 * hi;
      pl[row * 32 + nloc]      = (_Float16)u0.e[r];
      pl[row * 32 + 16 + nloc] = (_Float16)u1.e[r];
    }

    // rescale running O by alpha (O D-layout rows match r + 8*hi)
    for (int t = 0; t < 8; ++t) {
      V8F a_; a_.v = oacc[t];
      for (int r = 0; r < 8; ++r) a_.e[r] *= alpha[r];
      oacc[t] = a_.v;
    }

    __builtin_amdgcn_wave_barrier();           // order LDS store -> load
    const v16h pf = load_a_frag(pl, 32);

    // O += P @ V over all 128 channels (V tile [c][j] -> contiguous K=j)
    for (int t = 0; t < 8; ++t) {
      v16h vf = load_bt_frag(vt + (size_t)(t * 16) * 32, 32);
      oacc[t] = wmma16(pf, vf, oacc[t]);
    }

    // next chunk's DMA must be complete before anyone reads it
    if (wid == 0) __builtin_amdgcn_s_wait_tensorcnt(0);
    __syncthreads();
  }

  // finalize: divide by softmax denominator, store [B][N][C] f16
  for (int t = 0; t < 8; ++t) {
    V8F a_; a_.v = oacc[t];
    const int c = t * 16 + nloc;
    _Float16* dst = o16 + (size_t)b * NTOK * CH + c;
    for (int r = 0; r < 8; ++r) {
      const int n = n0 + r + 8 * hi;
      dst[(size_t)n * CH] = (_Float16)(a_.e[r] / lrow[r]);
    }
  }
}

// ---- Kernel 6: proj GEMM + bias + residual, output [B][C][N] f32 -----------

__global__ void __launch_bounds__(256) proj_kernel(const _Float16* __restrict__ o16,
                                                   const _Float16* __restrict__ w16p,
                                                   const float* __restrict__ proj_b,
                                                   const float* __restrict__ x,
                                                   float* __restrict__ out) {
  const int gw = blockIdx.x * (blockDim.x >> 5) + (threadIdx.x >> 5);
  const int ot = gw % 8;                  // CH/16 col tiles
  const int mt = gw / 8;                  // B*N/16 row tiles
  const int b  = mt >> 8;
  const int n0 = (mt & 255) << 4;
  const int o0 = ot << 4;
  const int lane = threadIdx.x & 31;
  const int nloc = lane & 15;
  const int hi   = lane >> 4;

  const _Float16* abase = o16 + ((size_t)b * NTOK + n0) * CH;
  v8f acc = {};
  for (int kk = 0; kk < 4; ++kk) {
    v16h a  = load_a_frag(abase + kk * 32, CH);
    v16h bf = load_bt_frag(w16p + (size_t)o0 * CH + kk * 32, CH);
    acc = wmma16(a, bf, acc);
  }
  const int o = o0 + nloc;
  const float bias = proj_b[o];
  V8F d; d.v = acc;
  const size_t base = ((size_t)b * CH + o) * NTOK + n0 + 8 * hi;
  for (int r = 0; r < 8; ++r)
    out[base + r] = x[base + r] + d.e[r] + bias;
}

// ---- Host launch -----------------------------------------------------------

extern "C" void kernel_launch(void* const* d_in, const int* in_sizes, int n_in,
                              void* d_out, int out_size, void* d_ws, size_t ws_size,
                              hipStream_t stream) {
  const float* x      = (const float*)d_in[0];
  const float* norm_w = (const float*)d_in[1];
  const float* norm_b = (const float*)d_in[2];
  const float* qkv_w  = (const float*)d_in[3];
  const float* qkv_b  = (const float*)d_in[4];
  const float* proj_w = (const float*)d_in[5];
  const float* proj_b = (const float*)d_in[6];
  float* out = (float*)d_out;

  char* ws = (char*)d_ws;
  size_t off = 0;
  auto carve = [&](size_t bytes) -> void* {
    void* p = ws + off;
    off += (bytes + 255) & ~(size_t)255;
    return p;
  };
  float*     stats = (float*)    carve((size_t)BATCH * GRP * 2 * sizeof(float));
  _Float16*  w16q  = (_Float16*) carve((size_t)3 * CH * CH * sizeof(_Float16));
  _Float16*  w16p  = (_Float16*) carve((size_t)CH * CH * sizeof(_Float16));
  const size_t act = (size_t)BATCH * NTOK * CH * sizeof(_Float16);   // 4 MB each
  _Float16*  h16   = (_Float16*) carve(act);
  _Float16*  q16   = (_Float16*) carve(act);
  _Float16*  k16   = (_Float16*) carve(act);
  _Float16*  v16   = (_Float16*) carve(act);
  _Float16*  o16   = (_Float16*) carve(act);

  gn_stats_kernel<<<BATCH * GRP, 256, 0, stream>>>(x, stats);
  gn_apply_kernel<<<(BATCH * CH * NTOK) / 256, 256, 0, stream>>>(x, norm_w, norm_b, stats, h16);
  cvt_w_kernel<<<(3 * CH * CH) / 256, 256, 0, stream>>>(qkv_w, proj_w, w16q, w16p);
  // QKV: (B*N/16) * (384/16) tiles, 8 waves per block
  qkv_gemm_kernel<<<(BATCH * (NTOK / 16) * 24) / 8, 256, 0, stream>>>(h16, w16q, qkv_b, q16, k16, v16);
  // Attention: 128 queries per block (8 waves), 32 blocks per batch
  attn_kernel<<<BATCH * (NTOK / 128), 256, 0, stream>>>(q16, k16, v16, o16);
  // Proj: (B*N/16) * (CH/16) tiles, 8 waves per block
  proj_kernel<<<(BATCH * (NTOK / 16) * (CH / 16)) / 8, 256, 0, stream>>>(o16, w16p, proj_b, x, out);
}
